// HyperbolicAttentionLayer_71545565216808
// MI455X (gfx1250) — compile-verified, matrix operand order checked by reading, not run on hardware
//
#include <hip/hip_runtime.h>
#include <hip/hip_bf16.h>

// ---------------------------------------------------------------------------
// Types for CDNA5 WMMA (wave32): bf16 A/B fragments (16 x bf16), f32 C/D (8 x f32)
// ---------------------------------------------------------------------------
typedef __attribute__((ext_vector_type(16))) __bf16        v16bf;
typedef __attribute__((ext_vector_type(8)))  float         v8f;
typedef __attribute__((ext_vector_type(4)))  unsigned int  u32x4;
typedef int v4i_g __attribute__((vector_size(16)));        // GCC-style, matches builtin

union FragAB { v16bf v; u32x4 q[2]; };

// CDNA5 async global->LDS copy path (ASYNCcnt), with portable fallback.
#if defined(__gfx1250__) && __has_builtin(__builtin_amdgcn_global_load_async_to_lds_b128)
#define HAVE_ASYNC_LDS 1
#endif

__device__ __forceinline__ void cp_b128(const unsigned short* g, unsigned short* l) {
#ifdef HAVE_ASYNC_LDS
  __builtin_amdgcn_global_load_async_to_lds_b128(
      (__attribute__((address_space(1))) v4i_g*)(unsigned short*)g,
      (__attribute__((address_space(3))) v4i_g*)l, 0, 0);
#else
  *(u32x4*)l = *(const u32x4*)g;
#endif
}

__device__ __forceinline__ void async_wait0() {
#ifdef HAVE_ASYNC_LDS
#if __has_builtin(__builtin_amdgcn_s_wait_asynccnt)
  __builtin_amdgcn_s_wait_asynccnt(0);
#else
  asm volatile("s_wait_asynccnt 0x0" ::: "memory");
#endif
#endif
}

__device__ __forceinline__ unsigned short f2bf(float f) {
  union { float f; unsigned int u; } v; v.f = f;
  unsigned int r = v.u + 0x7fffu + ((v.u >> 16) & 1u);   // round-to-nearest-even
  return (unsigned short)(r >> 16);
}

constexpr int B_  = 2;
constexpr int L_  = 2048;
constexpr int D_  = 1024;
constexpr int H_  = 16;
constexpr int HD_ = 64;
constexpr int M_  = B_ * L_;     // 4096 tokens

// ---------------------------------------------------------------------------
// fp32 -> bf16 conversion (weights)
// ---------------------------------------------------------------------------
__global__ void k_f32_to_bf16(const float* __restrict__ s,
                              unsigned short* __restrict__ d, int n) {
  int i = blockIdx.x * blockDim.x + threadIdx.x;
  int stride = gridDim.x * blockDim.x;
  for (; i < n; i += stride) d[i] = f2bf(s[i]);
}

__global__ void k_concat3(const float* __restrict__ a, const float* __restrict__ b,
                          const float* __restrict__ c, float* __restrict__ d) {
  int i = blockIdx.x * blockDim.x + threadIdx.x;
  if (i < D_)            d[i] = a[i];
  else if (i < 2 * D_)   d[i] = b[i - D_];
  else if (i < 3 * D_)   d[i] = c[i - 2 * D_];
}

// ---------------------------------------------------------------------------
// LayerNorm over D=1024, one block (256 thr) per row, bf16 output
// ---------------------------------------------------------------------------
__global__ __launch_bounds__(256) void k_layernorm_bf16(
    const float* __restrict__ x, const float* __restrict__ g,
    const float* __restrict__ bt, unsigned short* __restrict__ y) {
  int row = blockIdx.x, t = threadIdx.x;
  const float* xr = x + (size_t)row * D_;
  float v[4];
#pragma unroll
  for (int j = 0; j < 4; ++j) v[j] = xr[t + 256 * j];
  __shared__ float red[256];
  red[t] = v[0] + v[1] + v[2] + v[3];
  __syncthreads();
  for (int o = 128; o; o >>= 1) { if (t < o) red[t] += red[t + o]; __syncthreads(); }
  float mean = red[0] * (1.0f / D_);
  __syncthreads();
  float q = 0.f;
#pragma unroll
  for (int j = 0; j < 4; ++j) { float d = v[j] - mean; q += d * d; }
  red[t] = q;
  __syncthreads();
  for (int o = 128; o; o >>= 1) { if (t < o) red[t] += red[t + o]; __syncthreads(); }
  float rstd = rsqrtf(red[0] * (1.0f / D_) + 1e-5f);
#pragma unroll
  for (int j = 0; j < 4; ++j) {
    int c = t + 256 * j;
    y[(size_t)row * D_ + c] = f2bf((v[j] - mean) * rstd * g[c] + bt[c]);
  }
}

// ---------------------------------------------------------------------------
// Tiled bf16 WMMA GEMM: Y[M,N] = A[M,K] @ W[N,K]^T + bias
//   mode 0: f32 output     mode 1: GELU(erf) -> bf16 output
// Block tile 128x128, BK=32, 8 waves each own 64x32; double-buffered LDS,
// global->LDS staging via async-to-LDS when available.
// ---------------------------------------------------------------------------
#define BM 128
#define BN 128
#define BK 32

__global__ __launch_bounds__(256) void k_gemm_bf16(
    const unsigned short* __restrict__ A, const unsigned short* __restrict__ W,
    const float* __restrict__ bias, float* __restrict__ Yf,
    unsigned short* __restrict__ Yb, int M, int N, int K, int mode) {
  __shared__ __align__(16) unsigned short As[2][BM * BK];
  __shared__ __align__(16) unsigned short Ws[2][BN * BK];

  const int tid  = threadIdx.x;
  const int lane = tid & 31, wave = tid >> 5;
  const int wm   = wave >> 2, wn = wave & 3;      // 2x4 wave grid
  const int l15  = lane & 15, hf = lane >> 4;
  const int m0   = blockIdx.y * BM, n0 = blockIdx.x * BN;

  const int sr = tid >> 1;            // 0..127 (row within tile)
  const int sc = (tid & 1) * 16;      // 0 or 16 (K offset)

  v8f vzero = {};
  v8f acc[4][2];
#pragma unroll
  for (int i = 0; i < 4; ++i)
#pragma unroll
    for (int j = 0; j < 2; ++j) acc[i][j] = vzero;

  auto stage = [&](int b, int k0) {
    const unsigned short* ap = A + (size_t)(m0 + sr) * K + k0 + sc;
    cp_b128(ap,     &As[b][sr * BK + sc]);
    cp_b128(ap + 8, &As[b][sr * BK + sc + 8]);
    const unsigned short* wp = W + (size_t)(n0 + sr) * K + k0 + sc;
    cp_b128(wp,     &Ws[b][sr * BK + sc]);
    cp_b128(wp + 8, &Ws[b][sr * BK + sc + 8]);
  };

  const int nkt = K / BK;
  stage(0, 0);
  async_wait0();
  __syncthreads();

  for (int kt = 0; kt < nkt; ++kt) {
    int buf = kt & 1;
    if (kt + 1 < nkt) stage(buf ^ 1, (kt + 1) * BK);

    FragAB afr[4];
#pragma unroll
    for (int fm = 0; fm < 4; ++fm) {
      int row = wm * 64 + fm * 16 + l15;
      afr[fm].q[0] = *(const u32x4*)&As[buf][row * BK + hf * 8];
      afr[fm].q[1] = *(const u32x4*)&As[buf][row * BK + 16 + hf * 8];
    }
#pragma unroll
    for (int fn = 0; fn < 2; ++fn) {
      FragAB wfr;
      int wrow = wn * 32 + fn * 16 + l15;
      wfr.q[0] = *(const u32x4*)&Ws[buf][wrow * BK + hf * 8];
      wfr.q[1] = *(const u32x4*)&Ws[buf][wrow * BK + 16 + hf * 8];
#pragma unroll
      for (int fm = 0; fm < 4; ++fm)
        acc[fm][fn] = __builtin_amdgcn_wmma_f32_16x16x32_bf16(
            false, afr[fm].v, false, wfr.v, (short)0, acc[fm][fn], false, false);
    }
    async_wait0();
    __syncthreads();
  }

#pragma unroll
  for (int fm = 0; fm < 4; ++fm) {
#pragma unroll
    for (int fn = 0; fn < 2; ++fn) {
      int col = n0 + wn * 32 + fn * 16 + l15;
      float bcol = bias[col];
#pragma unroll
      for (int e = 0; e < 8; ++e) {
        int row = m0 + wm * 64 + fm * 16 + e + 8 * hf;
        float v = acc[fm][fn][e] + bcol;
        size_t idx = (size_t)row * N + col;
        if (mode == 0) Yf[idx] = v;
        else Yb[idx] = f2bf(0.5f * v * (1.0f + erff(v * 0.70710678118f)));
      }
    }
  }
}

// ---------------------------------------------------------------------------
// QKV post: Poincare projection per 64-d head vector for Q/K, plus ||.||^2;
// V is written transposed [B,H,hd,L] so PV B-fragments are contiguous.
// One 64-thread block per (which,b,h,l) vector.
// ---------------------------------------------------------------------------
__global__ __launch_bounds__(64) void k_qkv_post(
    const float* __restrict__ qkv, unsigned short* __restrict__ qb,
    unsigned short* __restrict__ kb, unsigned short* __restrict__ vt,
    float* __restrict__ q2, float* __restrict__ k2) {
  int vec = blockIdx.x;                      // 0 .. 3*B*H*L-1
  int which = vec / (B_ * H_ * L_);
  int rem = vec % (B_ * H_ * L_);
  int b = rem / (H_ * L_);
  int rem2 = rem % (H_ * L_);
  int h = rem2 / L_, l = rem2 % L_;
  int d = threadIdx.x;
  float v = qkv[((size_t)(b * L_ + l)) * (3 * D_) + which * D_ + h * HD_ + d];
  size_t bh = (size_t)(b * H_ + h);
  if (which < 2) {
    int lane = d & 31, wid = d >> 5;
    float ss = v * v;
#pragma unroll
    for (int o = 16; o; o >>= 1) ss += __shfl_xor(ss, o, 32);
    __shared__ float sh[2];
    if (lane == 0) sh[wid] = ss;
    __syncthreads();
    float n2 = sh[0] + sh[1];
    float n = sqrtf(n2);
    float scale = (n > 0.99999f) ? (0.99999f / fmaxf(n, 1e-12f)) : 1.0f;
    v *= scale;
    size_t dst = (bh * L_ + l) * HD_ + d;
    float n2p = n2 * scale * scale;
    if (which == 0) { qb[dst] = f2bf(v); if (d == 0) q2[bh * L_ + l] = n2p; }
    else            { kb[dst] = f2bf(v); if (d == 0) k2[bh * L_ + l] = n2p; }
  } else {
    vt[(bh * HD_ + d) * L_ + l] = f2bf(v);
  }
}

// ---------------------------------------------------------------------------
// Flash-style hyperbolic attention. Grid (L/128, B*H), 256 thr (8 waves).
// Wave owns 16 query rows; KV tiles of 64; scores = (2qk - q2 - k2)/temp,
// online softmax, P staged via wave-private LDS into A-fragment layout.
// ---------------------------------------------------------------------------
constexpr int QT = 128;
constexpr int KT = 64;

__global__ __launch_bounds__(256) void k_attention(
    const unsigned short* __restrict__ qb, const unsigned short* __restrict__ kb,
    const unsigned short* __restrict__ vt, const float* __restrict__ q2,
    const float* __restrict__ k2, unsigned short* __restrict__ attn_out) {
  __shared__ __align__(16) unsigned short Ks[KT * HD_];       // 8KB
  __shared__ __align__(16) unsigned short Vs[HD_ * KT];       // 8KB
  __shared__ __align__(16) unsigned short Ps[8 * 16 * KT];    // 16KB

  const int tid = threadIdx.x, lane = tid & 31, wv = tid >> 5;
  const int l15 = lane & 15, hf = lane >> 4;
  const int bh = blockIdx.y;
  const int b = bh >> 4, h = bh & 15;
  const int q0 = blockIdx.x * QT + wv * 16;
  const float invT = 0.125f;                 // 1/sqrt(hd=64)

  FragAB qf[2];
  {
    const unsigned short* qrow = qb + ((size_t)bh * L_ + q0 + l15) * HD_;
#pragma unroll
    for (int kk = 0; kk < 2; ++kk) {
      qf[kk].q[0] = *(const u32x4*)&qrow[kk * 32 + hf * 8];
      qf[kk].q[1] = *(const u32x4*)&qrow[kk * 32 + 16 + hf * 8];
    }
  }
  float q2r[8];
#pragma unroll
  for (int e = 0; e < 8; ++e) q2r[e] = q2[(size_t)bh * L_ + q0 + e + 8 * hf];

  v8f vzero = {};
  v8f ov[4];
  float mrow[8], lrow[8];
#pragma unroll
  for (int j = 0; j < 4; ++j) ov[j] = vzero;
#pragma unroll
  for (int e = 0; e < 8; ++e) { mrow[e] = -1e30f; lrow[e] = 0.f; }

  for (int kv0 = 0; kv0 < L_; kv0 += KT) {
    __syncthreads();
    {   // stage K tile [KT][HD] and V^T tile [HD][KT]
      int r = tid >> 2, c = (tid & 3) * 16;
      const unsigned short* ksrc = kb + ((size_t)bh * L_ + kv0 + r) * HD_ + c;
      cp_b128(ksrc,     &Ks[r * HD_ + c]);
      cp_b128(ksrc + 8, &Ks[r * HD_ + c + 8]);
      const unsigned short* vsrc = vt + ((size_t)bh * HD_ + r) * L_ + kv0 + c;
      cp_b128(vsrc,     &Vs[r * KT + c]);
      cp_b128(vsrc + 8, &Vs[r * KT + c + 8]);
    }
    async_wait0();
    __syncthreads();

    // S = Q @ K^T  (4 n-frags of 16, K=64 -> 2 wmma each)
    v8f sf[4];
    float k2v[4];
#pragma unroll
    for (int nf = 0; nf < 4; ++nf) {
      v8f s = vzero;
#pragma unroll
      for (int kk = 0; kk < 2; ++kk) {
        FragAB bf;
        int krow = nf * 16 + l15;
        bf.q[0] = *(const u32x4*)&Ks[krow * HD_ + kk * 32 + hf * 8];
        bf.q[1] = *(const u32x4*)&Ks[krow * HD_ + kk * 32 + 16 + hf * 8];
        s = __builtin_amdgcn_wmma_f32_16x16x32_bf16(
            false, qf[kk].v, false, bf.v, (short)0, s, false, false);
      }
      sf[nf] = s;
      k2v[nf] = k2[(size_t)bh * L_ + kv0 + nf * 16 + l15];
    }

    // online softmax; write P (bf16) to wave-private LDS region
#pragma unroll
    for (int e = 0; e < 8; ++e) {
      float p[4];
      float mx = -1e30f;
#pragma unroll
      for (int nf = 0; nf < 4; ++nf) {
        float val = (2.0f * sf[nf][e] - q2r[e] - k2v[nf]) * invT;
        p[nf] = val;
        mx = fmaxf(mx, val);
      }
#pragma unroll
      for (int o = 1; o < 16; o <<= 1) mx = fmaxf(mx, __shfl_xor(mx, o, 32));
      float mnew = fmaxf(mrow[e], mx);
      float alpha = __expf(mrow[e] - mnew);
      mrow[e] = mnew;
      float sum = 0.f;
      int prow = (wv * 16 + e + 8 * hf) * KT;
#pragma unroll
      for (int nf = 0; nf < 4; ++nf) {
        float pe = __expf(p[nf] - mnew);
        sum += pe;
        Ps[prow + nf * 16 + l15] = f2bf(pe);
      }
#pragma unroll
      for (int o = 1; o < 16; o <<= 1) sum += __shfl_xor(sum, o, 32);
      lrow[e] = lrow[e] * alpha + sum;
#pragma unroll
      for (int j = 0; j < 4; ++j) ov[j][e] *= alpha;
    }

    // O += P @ V   (Kv=64 -> 2 steps of 32; 4 output n-frags over hd)
#pragma unroll
    for (int nk = 0; nk < 2; ++nk) {
      FragAB pf;
      int prow = (wv * 16 + l15) * KT;
      pf.q[0] = *(const u32x4*)&Ps[prow + nk * 32 + hf * 8];
      pf.q[1] = *(const u32x4*)&Ps[prow + nk * 32 + 16 + hf * 8];
#pragma unroll
      for (int nfo = 0; nfo < 4; ++nfo) {
        FragAB vf;
        int drow = nfo * 16 + l15;
        vf.q[0] = *(const u32x4*)&Vs[drow * KT + nk * 32 + hf * 8];
        vf.q[1] = *(const u32x4*)&Vs[drow * KT + nk * 32 + 16 + hf * 8];
        ov[nfo] = __builtin_amdgcn_wmma_f32_16x16x32_bf16(
            false, pf.v, false, vf.v, (short)0, ov[nfo], false, false);
      }
    }
  }

#pragma unroll
  for (int nfo = 0; nfo < 4; ++nfo) {
#pragma unroll
    for (int e = 0; e < 8; ++e) {
      int l = q0 + e + 8 * hf;
      int d = nfo * 16 + l15;
      float val = ov[nfo][e] / fmaxf(lrow[e], 1e-20f);
      attn_out[((size_t)(b * L_ + l)) * D_ + h * HD_ + d] = f2bf(val);
    }
  }
}

// ---------------------------------------------------------------------------
// y = project(x + add) over full D row; optionally also LayerNorm -> bf16
// ---------------------------------------------------------------------------
__global__ __launch_bounds__(256) void k_res_proj_ln(
    const float* __restrict__ x, const float* __restrict__ add,
    const float* __restrict__ g, const float* __restrict__ bt,
    float* __restrict__ x1, unsigned short* __restrict__ xn2) {
  int row = blockIdx.x, t = threadIdx.x;
  const float* xr = x + (size_t)row * D_;
  const float* ar = add + (size_t)row * D_;
  float v[4];
#pragma unroll
  for (int j = 0; j < 4; ++j) v[j] = xr[t + 256 * j] + ar[t + 256 * j];
  __shared__ float red[256];
  red[t] = v[0] * v[0] + v[1] * v[1] + v[2] * v[2] + v[3] * v[3];
  __syncthreads();
  for (int o = 128; o; o >>= 1) { if (t < o) red[t] += red[t + o]; __syncthreads(); }
  float n = sqrtf(red[0]);
  float scale = (n > 0.99999f) ? (0.99999f / fmaxf(n, 1e-12f)) : 1.0f;
  __syncthreads();
#pragma unroll
  for (int j = 0; j < 4; ++j) {
    v[j] *= scale;
    x1[(size_t)row * D_ + t + 256 * j] = v[j];
  }
  if (xn2) {
    red[t] = v[0] + v[1] + v[2] + v[3];
    __syncthreads();
    for (int o = 128; o; o >>= 1) { if (t < o) red[t] += red[t + o]; __syncthreads(); }
    float mean = red[0] * (1.0f / D_);
    __syncthreads();
    float q = 0.f;
#pragma unroll
    for (int j = 0; j < 4; ++j) { float d = v[j] - mean; q += d * d; }
    red[t] = q;
    __syncthreads();
    for (int o = 128; o; o >>= 1) { if (t < o) red[t] += red[t + o]; __syncthreads(); }
    float rstd = rsqrtf(red[0] * (1.0f / D_) + 1e-5f);
#pragma unroll
    for (int j = 0; j < 4; ++j) {
      int c = t + 256 * j;
      xn2[(size_t)row * D_ + c] = f2bf((v[j] - mean) * rstd * g[c] + bt[c]);
    }
  }
}

// ---------------------------------------------------------------------------
extern "C" void kernel_launch(void* const* d_in, const int* in_sizes, int n_in,
                              void* d_out, int out_size, void* d_ws, size_t ws_size,
                              hipStream_t stream) {
  (void)in_sizes; (void)n_in; (void)out_size; (void)ws_size;
  const float* x   = (const float*)d_in[0];
  const float* wq  = (const float*)d_in[1];  const float* bq  = (const float*)d_in[2];
  const float* wk  = (const float*)d_in[3];  const float* bk  = (const float*)d_in[4];
  const float* wvp = (const float*)d_in[5];  const float* bv  = (const float*)d_in[6];
  const float* wo  = (const float*)d_in[7];  const float* bo  = (const float*)d_in[8];
  const float* g1  = (const float*)d_in[9];  const float* b1  = (const float*)d_in[10];
  const float* g2  = (const float*)d_in[11]; const float* b2  = (const float*)d_in[12];
  const float* w1  = (const float*)d_in[13]; const float* bf1 = (const float*)d_in[14];
  const float* w2  = (const float*)d_in[15]; const float* bf2 = (const float*)d_in[16];
  float* out = (float*)d_out;

  char* ws = (char*)d_ws;
  size_t off = 0;
  auto take = [&](size_t bytes) -> void* {
    void* p = ws + off;
    off += (bytes + 255) & ~(size_t)255;
    return p;
  };
  unsigned short* xn    = (unsigned short*)take((size_t)M_ * D_ * 2);
  unsigned short* wqkvb = (unsigned short*)take((size_t)3 * D_ * D_ * 2);
  unsigned short* wob   = (unsigned short*)take((size_t)D_ * D_ * 2);
  unsigned short* w1b   = (unsigned short*)take((size_t)4 * D_ * D_ * 2);
  unsigned short* w2b   = (unsigned short*)take((size_t)4 * D_ * D_ * 2);
  float*          bqkv  = (float*)take((size_t)3 * D_ * 4);
  float*          qkvf  = (float*)take((size_t)M_ * 3 * D_ * 4);
  unsigned short* qb    = (unsigned short*)take((size_t)M_ * D_ * 2);
  unsigned short* kb    = (unsigned short*)take((size_t)M_ * D_ * 2);
  unsigned short* vt    = (unsigned short*)take((size_t)M_ * D_ * 2);
  float*          q2    = (float*)take((size_t)B_ * H_ * L_ * 4);
  float*          k2    = (float*)take((size_t)B_ * H_ * L_ * 4);
  unsigned short* attnb = (unsigned short*)take((size_t)M_ * D_ * 2);
  float*          aproj = (float*)take((size_t)M_ * D_ * 4);
  float*          x1    = (float*)take((size_t)M_ * D_ * 4);
  unsigned short* xn2   = (unsigned short*)take((size_t)M_ * D_ * 2);
  unsigned short* hb    = (unsigned short*)take((size_t)M_ * 4 * D_ * 2);
  float*          y2    = (float*)take((size_t)M_ * D_ * 4);

  const int DD = D_ * D_;
  // 1) weights -> bf16 (wq|wk|wv fused for one QKV GEMM)
  k_f32_to_bf16<<<2048, 256, 0, stream>>>(wq,  wqkvb,           DD);
  k_f32_to_bf16<<<2048, 256, 0, stream>>>(wk,  wqkvb + DD,      DD);
  k_f32_to_bf16<<<2048, 256, 0, stream>>>(wvp, wqkvb + 2 * DD,  DD);
  k_f32_to_bf16<<<2048, 256, 0, stream>>>(wo,  wob,             DD);
  k_f32_to_bf16<<<4096, 256, 0, stream>>>(w1,  w1b,         4 * DD);
  k_f32_to_bf16<<<4096, 256, 0, stream>>>(w2,  w2b,         4 * DD);
  k_concat3<<<12, 256, 0, stream>>>(bq, bk, bv, bqkv);

  // 2) LN1 -> bf16
  k_layernorm_bf16<<<M_, 256, 0, stream>>>(x, g1, b1, xn);

  // 3) fused QKV GEMM: [4096,1024] @ [3072,1024]^T
  k_gemm_bf16<<<dim3(3 * D_ / BN, M_ / BM), 256, 0, stream>>>(
      xn, wqkvb, bqkv, qkvf, nullptr, M_, 3 * D_, D_, 0);

  // 4) Poincare projection + head split (V transposed)
  k_qkv_post<<<3 * B_ * H_ * L_, 64, 0, stream>>>(qkvf, qb, kb, vt, q2, k2);

  // 5) attention
  k_attention<<<dim3(L_ / QT, B_ * H_), 256, 0, stream>>>(qb, kb, vt, q2, k2, attnb);

  // 6) output projection
  k_gemm_bf16<<<dim3(D_ / BN, M_ / BM), 256, 0, stream>>>(
      attnb, wob, bo, aproj, nullptr, M_, D_, D_, 0);

  // 7) residual + project + LN2
  k_res_proj_ln<<<M_, 256, 0, stream>>>(x, aproj, g2, b2, x1, xn2);

  // 8) FFN up (GELU -> bf16)
  k_gemm_bf16<<<dim3(4 * D_ / BN, M_ / BM), 256, 0, stream>>>(
      xn2, w1b, bf1, nullptr, hb, M_, 4 * D_, D_, 1);

  // 9) FFN down
  k_gemm_bf16<<<dim3(D_ / BN, M_ / BM), 256, 0, stream>>>(
      hb, w2b, bf2, y2, nullptr, M_, D_, 4 * D_, 0);

  // 10) final residual + project -> f32 output
  k_res_proj_ln<<<M_, 256, 0, stream>>>(x1, y2, nullptr, nullptr, out, nullptr);
}
